// Block_8014408975059
// MI455X (gfx1250) — compile-verified, matrix operand order checked by reading, not run on hardware
//
#include <hip/hip_runtime.h>
#include <hip/hip_bf16.h>

// ---------------------------------------------------------------------------
// Types / helpers
// ---------------------------------------------------------------------------
typedef __attribute__((ext_vector_type(16))) __bf16 v16bf;
typedef __attribute__((ext_vector_type(8)))  float  v8f;

#define DEV __device__ __forceinline__

DEV __bf16 f2bf(float f) {
    unsigned u = __builtin_bit_cast(unsigned, f);
    unsigned r = u + 0x7fffu + ((u >> 16) & 1u);   // round-to-nearest-even
    unsigned short h = (unsigned short)(r >> 16);
    return __builtin_bit_cast(__bf16, h);
}

// CDNA5 async global->LDS copy, 16B per lane (tracked by ASYNCcnt).
// %0 = per-lane LDS byte address (flat LDS addr == addr[31:0]), %1 = 64b vaddr
DEV void async_ld_b128(unsigned lds_byte, const void* gaddr) {
    asm volatile("global_load_async_to_lds_b128 %0, %1, off"
                 :: "v"(lds_byte), "v"(gaddr) : "memory");
}
#define WAIT_ASYNC(N) asm volatile("s_wait_asynccnt " #N ::: "memory")

DEV unsigned lds_addr(const void* p) { return (unsigned)(uintptr_t)p; }

// Fragment gather matching CDNA5 16-bit A/B 16x32 VGPR layout:
//   lanes 0-15 : row = lane,    elems 0-7 -> K 0-7,   elems 8-15 -> K 16-23
//   lanes 16-31: row = lane-16, elems 0-7 -> K 8-15,  elems 8-15 -> K 24-31
DEV v16bf load_frag_g(const __bf16* base, int ld, int lane) {
    int row  = lane & 15;
    int koff = (lane >> 4) << 3;
    const __bf16* p = base + (size_t)row * ld + koff;
    v16bf f;
#pragma unroll
    for (int i = 0; i < 8; ++i) { f[i] = p[i]; f[i + 8] = p[i + 16]; }
    return f;
}

template <int LD>
DEV v16bf load_frag_s(const __bf16* base, int lane) {
    int row  = lane & 15;
    int koff = (lane >> 4) << 3;
    const __bf16* p = base + row * LD + koff;
    v16bf f;
#pragma unroll
    for (int i = 0; i < 8; ++i) { f[i] = p[i]; f[i + 8] = p[i + 16]; }
    return f;
}

DEV v8f wmma_bf16(v16bf a, v16bf b, v8f c) {
    return __builtin_amdgcn_wmma_f32_16x16x32_bf16(false, a, false, b,
                                                   (short)0, c, false, false);
}

// ---------------------------------------------------------------------------
// Weight convert + transpose: fp32 W[K,N] -> bf16 Wt[N,K]
// ---------------------------------------------------------------------------
__global__ __launch_bounds__(256)
void wconv_kernel(const float* __restrict__ W, __bf16* __restrict__ Wt,
                  int K, int N) {
    int idx = blockIdx.x * 256 + threadIdx.x;
    if (idx >= K * N) return;
    int kk = idx / N, n = idx - kk * N;
    Wt[(size_t)n * K + kk] = f2bf(W[idx]);
}

// ---------------------------------------------------------------------------
// LayerNorm: fp32 x[R,1024] -> bf16 out[R,1024]
// ---------------------------------------------------------------------------
__global__ __launch_bounds__(256)
void ln_kernel(const float* __restrict__ x, const float* __restrict__ g,
               const float* __restrict__ bta, __bf16* __restrict__ out) {
    const int C = 1024;
    int row = blockIdx.x;
    int t   = threadIdx.x;
    const float* xr = x + (size_t)row * C;
    float v[4];
    float s = 0.f, ss = 0.f;
#pragma unroll
    for (int i = 0; i < 4; ++i) {
        float f = xr[t + i * 256];
        v[i] = f; s += f; ss += f * f;
    }
    __shared__ float rs[256], rss[256];
    rs[t] = s; rss[t] = ss;
    __syncthreads();
    for (int stp = 128; stp > 0; stp >>= 1) {
        if (t < stp) { rs[t] += rs[t + stp]; rss[t] += rss[t + stp]; }
        __syncthreads();
    }
    float mu  = rs[0] * (1.0f / C);
    float var = rss[0] * (1.0f / C) - mu * mu;
    float inv = rsqrtf(var + 1e-5f);
#pragma unroll
    for (int i = 0; i < 4; ++i) {
        int c = t + i * 256;
        out[(size_t)row * C + c] = f2bf((v[i] - mu) * inv * g[c] + bta[c]);
    }
}

// ---------------------------------------------------------------------------
// WMMA GEMM: out = act(A[R,K] @ Wt[N,K]^T + bias) (+residual)
// 256 thr = 8 waves (2Mx4N). Block tile 128x256; wave tile 64x64
// (16 v_wmma per K-step vs 8 LDS fragment gathers).
// Double-buffered LDS fed by async global->LDS copies (ASYNCcnt pipelined).
// MODE 0: bf16 row-major | 1: bf16 per-head transposed (for V) |
// MODE 2: fp32 residual  | 3: bf16 exact GELU
// ---------------------------------------------------------------------------
template <int K, int N, int MODE>
__global__ __launch_bounds__(256)
void gemm_kernel(const __bf16* __restrict__ A, const __bf16* __restrict__ Wt,
                 const float* __restrict__ bias, const float* __restrict__ res,
                 void* __restrict__ outv) {
    __shared__ __attribute__((aligned(16))) __bf16 As[2][128 * 40];
    __shared__ __attribute__((aligned(16))) __bf16 Bs[2][256 * 40];
    int tid  = threadIdx.x;
    int lane = tid & 31, wid = tid >> 5;
    int wm = wid >> 2, wn = wid & 3;          // 2(M) x 4(N) waves
    size_t mBase = (size_t)blockIdx.y * 128;
    size_t nBase = (size_t)blockIdx.x * 256;
    const __bf16* Ab = A  + mBase * K;
    const __bf16* Bb = Wt + nBase * K;
    int ar = tid >> 1, ak = (tid & 1) * 16;   // A: 128 rows x 32 k, 16 elem/thr

    v8f acc[4][4] = {};

    auto issue = [&](int p, int kt) {
        const __bf16* ga = Ab + (size_t)ar * K + kt + ak;
        unsigned la = lds_addr(&As[p][ar * 40 + ak]);
        async_ld_b128(la,      ga);
        async_ld_b128(la + 16, ga + 8);
        const __bf16* gb = Bb + (size_t)tid * K + kt;   // B: 256 rows, 32 k/thr
        unsigned lb = lds_addr(&Bs[p][tid * 40]);
        async_ld_b128(lb,      gb);
        async_ld_b128(lb + 16, gb + 8);
        async_ld_b128(lb + 32, gb + 16);
        async_ld_b128(lb + 48, gb + 24);
    };

    issue(0, 0);
    const int nIt = K / 32;
    for (int it = 0; it < nIt; ++it) {
        int p = it & 1;
        if (it + 1 < nIt) { issue(1 - p, (it + 1) * 32); WAIT_ASYNC(6); }
        else              { WAIT_ASYNC(0); }
        __syncthreads();
        v16bf af[4];
#pragma unroll
        for (int i = 0; i < 4; ++i)
            af[i] = load_frag_s<40>(&As[p][(wm * 64 + i * 16) * 40], lane);
#pragma unroll
        for (int j = 0; j < 4; ++j) {
            v16bf bfj = load_frag_s<40>(&Bs[p][(wn * 64 + j * 16) * 40], lane);
#pragma unroll
            for (int i = 0; i < 4; ++i)
                acc[i][j] = wmma_bf16(af[i], bfj, acc[i][j]);
        }
        __syncthreads();
    }

    int hlf = lane >> 4, n15 = lane & 15;
#pragma unroll
    for (int i = 0; i < 4; ++i)
#pragma unroll
    for (int j = 0; j < 4; ++j) {
        int gn = (int)nBase + wn * 64 + j * 16 + n15;
        float bia = bias[gn];
#pragma unroll
        for (int r = 0; r < 8; ++r) {
            int gm = (int)mBase + wm * 64 + i * 16 + hlf * 8 + r;
            float v = acc[i][j][r] + bia;
            if (MODE == 3) v = 0.5f * v * (1.0f + erff(v * 0.70710678118654752f));
            if (MODE == 0 || MODE == 3) {
                ((__bf16*)outv)[(size_t)gm * N + gn] = f2bf(v);
            } else if (MODE == 2) {
                size_t idx = (size_t)gm * N + gn;
                ((float*)outv)[idx] = res[idx] + v;
            } else { // MODE == 1: V stored as Vt[b][h][d][t]
                int bb = gm >> 11, t = gm & 2047;
                int hh = gn >> 6,  d = gn & 63;
                ((__bf16*)outv)[(((size_t)bb * 16 + hh) * 64 + d) * 2048 + t] = f2bf(v);
            }
        }
    }
}

// ---------------------------------------------------------------------------
// Flash attention (causal). 128 thr = 4 waves; each wave owns 16 q rows.
// K/V tiles for each 32-key block are async-staged once into double-buffered
// LDS and shared by all 4 waves (uniform trip count; masked tail blocks are
// numerically inert). Q bf16 [B*T,C] head-strided; V pre-transposed [B,H,64,T].
// ---------------------------------------------------------------------------
__global__ __launch_bounds__(128)
void attn_kernel(const __bf16* __restrict__ Q, const __bf16* __restrict__ Km,
                 const __bf16* __restrict__ Vt, __bf16* __restrict__ O) {
    const int C = 1024, T = 2048, D = 64, H = 16;
    int tid  = threadIdx.x;
    int lane = tid & 31;
    int wid  = tid >> 5;
    int qblk = blockIdx.x, h = blockIdx.y, b = blockIdx.z;
    int qbase = qblk * 64 + wid * 16;
    int hlf = lane >> 4, n15 = lane & 15;

    __shared__ __attribute__((aligned(16))) __bf16 Ks[2][32 * 72];
    __shared__ __attribute__((aligned(16))) __bf16 Vs[2][64 * 40];
    __shared__ __attribute__((aligned(16))) __bf16 Ps[4][16 * 40];
    __bf16* pw = Ps[wid];

    const __bf16* qbp = Q + ((size_t)(b * T + qbase)) * C + h * D;
    v16bf qf0 = load_frag_g(qbp,      C, lane);
    v16bf qf1 = load_frag_g(qbp + 32, C, lane);
    const __bf16* kbp = Km + ((size_t)b * T) * C + h * D;
    const __bf16* vbp = Vt + ((size_t)b * H + h) * D * (size_t)T;

    int kr = tid >> 2, ksg = (tid & 3) * 16;  // K tile: 32 rows x 64 d
    int vr = tid >> 1, vsg = (tid & 1) * 16;  // V tile: 64 rows x 32 keys
    auto issue = [&](int p, int kb) {
        const __bf16* gk = kbp + (size_t)(kb * 32 + kr) * C + ksg;
        unsigned lk = lds_addr(&Ks[p][kr * 72 + ksg]);
        async_ld_b128(lk,      gk);
        async_ld_b128(lk + 16, gk + 8);
        const __bf16* gv = vbp + (size_t)vr * T + kb * 32 + vsg;
        unsigned lv = lds_addr(&Vs[p][vr * 40 + vsg]);
        async_ld_b128(lv,      gv);
        async_ld_b128(lv + 16, gv + 8);
    };

    v8f oacc[4] = {};
    float mrow[8], lrow[8];
#pragma unroll
    for (int r = 0; r < 8; ++r) { mrow[r] = -3.0e38f; lrow[r] = 0.f; }

    const int kbEnd = (qblk * 64 + 95) >> 5;  // uniform across the block
    issue(0, 0);
    for (int kb = 0; kb < kbEnd; ++kb) {
        int p = kb & 1;
        if (kb + 1 < kbEnd) { issue(1 - p, kb + 1); WAIT_ASYNC(4); }
        else                { WAIT_ASYNC(0); }
        __syncthreads();

        v8f s0 = {}, s1 = {};
        {
            v16bf k00 = load_frag_s<72>(&Ks[p][0],        lane);
            v16bf k01 = load_frag_s<72>(&Ks[p][32],       lane);
            s0 = wmma_bf16(qf0, k00, s0);
            s0 = wmma_bf16(qf1, k01, s0);
            v16bf k10 = load_frag_s<72>(&Ks[p][16 * 72],      lane);
            v16bf k11 = load_frag_s<72>(&Ks[p][16 * 72 + 32], lane);
            s1 = wmma_bf16(qf0, k10, s1);
            s1 = wmma_bf16(qf1, k11, s1);
        }
#pragma unroll
        for (int r = 0; r < 8; ++r) {
            int qrow = qbase + r + 8 * hlf;
            int kc0  = kb * 32 + n15;
            float a = s0[r] * 0.125f;            // 1/sqrt(64)
            float c = s1[r] * 0.125f;
            if (kc0      > qrow) a = -3.0e38f;
            if (kc0 + 16 > qrow) c = -3.0e38f;
            float mx = fmaxf(a, c);
#pragma unroll
            for (int m = 1; m < 16; m <<= 1)
                mx = fmaxf(mx, __shfl_xor(mx, m, 32));
            float mnew  = fmaxf(mrow[r], mx);
            float alpha = __expf(mrow[r] - mnew);
            float p0 = __expf(a - mnew);
            float p1 = __expf(c - mnew);
            float ps = p0 + p1;
#pragma unroll
            for (int m = 1; m < 16; m <<= 1)
                ps += __shfl_xor(ps, m, 32);
            lrow[r] = lrow[r] * alpha + ps;
            mrow[r] = mnew;
            pw[(r + 8 * hlf) * 40 + n15]      = f2bf(p0);
            pw[(r + 8 * hlf) * 40 + 16 + n15] = f2bf(p1);
#pragma unroll
            for (int j = 0; j < 4; ++j) oacc[j][r] *= alpha;
        }
        v16bf pf = load_frag_s<40>(pw, lane);
#pragma unroll
        for (int j = 0; j < 4; ++j) {
            v16bf vf = load_frag_s<40>(&Vs[p][(j * 16) * 40], lane);
            oacc[j] = wmma_bf16(pf, vf, oacc[j]);
        }
        __syncthreads();
    }
#pragma unroll
    for (int j = 0; j < 4; ++j)
#pragma unroll
    for (int r = 0; r < 8; ++r) {
        int grow = b * T + qbase + r + 8 * hlf;
        int gcol = h * D + j * 16 + n15;
        O[(size_t)grow * C + gcol] = f2bf(oacc[j][r] / lrow[r]);
    }
}

// ---------------------------------------------------------------------------
// Host launch
// ---------------------------------------------------------------------------
extern "C" void kernel_launch(void* const* d_in, const int* in_sizes, int n_in,
                              void* d_out, int out_size, void* d_ws, size_t ws_size,
                              hipStream_t stream) {
    (void)in_sizes; (void)n_in; (void)out_size; (void)ws_size;
    const float* x    = (const float*)d_in[0];
    const float* ln1g = (const float*)d_in[1];
    const float* ln1b = (const float*)d_in[2];
    const float* Wq   = (const float*)d_in[3];
    const float* bq   = (const float*)d_in[4];
    const float* Wk   = (const float*)d_in[5];
    const float* bk   = (const float*)d_in[6];
    const float* Wv   = (const float*)d_in[7];
    const float* bv   = (const float*)d_in[8];
    const float* Wo   = (const float*)d_in[9];
    const float* bo   = (const float*)d_in[10];
    const float* ln2g = (const float*)d_in[11];
    const float* ln2b = (const float*)d_in[12];
    const float* W1   = (const float*)d_in[13];
    const float* b1   = (const float*)d_in[14];
    const float* W2   = (const float*)d_in[15];
    const float* b2   = (const float*)d_in[16];

    const size_t R = 8192;   // B*T
    char* ws = (char*)d_ws;
    size_t off = 0;
    auto alloc = [&](size_t bytes) -> void* {
        void* p = ws + off;
        off += (bytes + 255) & ~(size_t)255;
        return p;
    };
    __bf16* wtq = (__bf16*)alloc((size_t)1024 * 1024 * 2);
    __bf16* wtk = (__bf16*)alloc((size_t)1024 * 1024 * 2);
    __bf16* wtv = (__bf16*)alloc((size_t)1024 * 1024 * 2);
    __bf16* wto = (__bf16*)alloc((size_t)1024 * 1024 * 2);
    __bf16* wt1 = (__bf16*)alloc((size_t)4096 * 1024 * 2);
    __bf16* wt2 = (__bf16*)alloc((size_t)4096 * 1024 * 2);
    __bf16* hbuf  = (__bf16*)alloc(R * 1024 * 2);
    __bf16* qbuf  = (__bf16*)alloc(R * 1024 * 2);
    __bf16* kbuf  = (__bf16*)alloc(R * 1024 * 2);
    __bf16* vtbuf = (__bf16*)alloc(R * 1024 * 2);
    __bf16* abuf  = (__bf16*)alloc(R * 1024 * 2);
    float*  x2    = (float*) alloc(R * 1024 * 4);
    __bf16* h2buf = (__bf16*)alloc(R * 1024 * 2);
    __bf16* mbuf  = (__bf16*)alloc(R * 4096 * 2);

    // Weight convert+transpose to bf16 [N,K]
    wconv_kernel<<<(1024 * 1024 + 255) / 256, 256, 0, stream>>>(Wq, wtq, 1024, 1024);
    wconv_kernel<<<(1024 * 1024 + 255) / 256, 256, 0, stream>>>(Wk, wtk, 1024, 1024);
    wconv_kernel<<<(1024 * 1024 + 255) / 256, 256, 0, stream>>>(Wv, wtv, 1024, 1024);
    wconv_kernel<<<(1024 * 1024 + 255) / 256, 256, 0, stream>>>(Wo, wto, 1024, 1024);
    wconv_kernel<<<(1024 * 4096 + 255) / 256, 256, 0, stream>>>(W1, wt1, 1024, 4096);
    wconv_kernel<<<(4096 * 1024 + 255) / 256, 256, 0, stream>>>(W2, wt2, 4096, 1024);

    // LN1 -> h (bf16)
    ln_kernel<<<(int)R, 256, 0, stream>>>(x, ln1g, ln1b, hbuf);

    // QKV projections
    gemm_kernel<1024, 1024, 0><<<dim3(4, 64), 256, 0, stream>>>(hbuf, wtq, bq, nullptr, qbuf);
    gemm_kernel<1024, 1024, 0><<<dim3(4, 64), 256, 0, stream>>>(hbuf, wtk, bk, nullptr, kbuf);
    gemm_kernel<1024, 1024, 1><<<dim3(4, 64), 256, 0, stream>>>(hbuf, wtv, bv, nullptr, vtbuf);

    // Causal flash attention
    attn_kernel<<<dim3(32, 16, 4), 128, 0, stream>>>(qbuf, kbuf, vtbuf, abuf);

    // Out projection + residual -> x2 (fp32)
    gemm_kernel<1024, 1024, 2><<<dim3(4, 64), 256, 0, stream>>>(abuf, wto, bo, x, x2);

    // LN2 -> h2
    ln_kernel<<<(int)R, 256, 0, stream>>>(x2, ln2g, ln2b, h2buf);

    // MLP: GELU(h2 @ W1 + b1) -> m ; x2 + m @ W2 + b2 -> d_out (fp32)
    gemm_kernel<1024, 4096, 3><<<dim3(16, 64), 256, 0, stream>>>(h2buf, wt1, b1, nullptr, mbuf);
    gemm_kernel<4096, 1024, 2><<<dim3(4, 64), 256, 0, stream>>>(mbuf, wt2, b2, x2, (float*)d_out);
}
